// LoFTREncoderLayer_56006373540107
// MI455X (gfx1250) — matrix operand order repaired and checked by reading, not compiled, not run on hardware
//
#include <hip/hip_runtime.h>
#include <math.h>

typedef __attribute__((ext_vector_type(16))) _Float16 v16h;
typedef __attribute__((ext_vector_type(8)))  _Float16 v8h;
typedef __attribute__((ext_vector_type(8)))  float    v8f;
typedef int vi4 __attribute__((vector_size(16)));   // matches builtin param type

#define DEV __device__ __forceinline__

// ---------------- problem constants ----------------
static constexpr int   NB = 4, LL = 4096, SSQ = 4096, HH = 8, DD = 64;
static constexpr int   TOK = NB * LL * HH;          // 131072 tokens
static constexpr float EPS_ATTN = 1e-6f;
static constexpr float EPS_LN   = 1e-5f;
static constexpr float INV_S    = 1.0f / (float)SSQ;

// ---------------- workspace layout (bytes) ----------------
static constexpr size_t WQH = 0;             // 64x64   f16
static constexpr size_t WKH = 8192;          // 64x64   f16
static constexpr size_t WVH = 16384;         // 64x64   f16
static constexpr size_t WMH = 24576;         // 64x64   f16
static constexpr size_t W1H = 32768;         // 128x128 f16
static constexpr size_t W2H = 65536;         // 64x128  f16
static constexpr size_t KVO = 81920;         // 32*64*64 f32 accumulators
static constexpr size_t KSO = 606208;        // 32*64    f32 accumulators (contiguous after KVO)
static constexpr size_t QFO = 1u << 20;      // TOK*64 f16  (elu(q)+1)
static constexpr size_t KFO = QFO + (size_t)TOK * DD * 2;   // elu(k)+1
static constexpr size_t VFO = KFO + (size_t)TOK * DD * 2;   // v / S

// ---------------- gfx1250 async copy to LDS (ASYNCcnt path) ----------------
#if defined(__gfx1250__) && __has_builtin(__builtin_amdgcn_global_load_async_to_lds_b128)
#define HAVE_ASYNC_B128 1
#else
#define HAVE_ASYNC_B128 0
#endif

DEV void async_cp_b128(const _Float16* g, _Float16* l) {
#if HAVE_ASYNC_B128
  __builtin_amdgcn_global_load_async_to_lds_b128(
      (__attribute__((address_space(1))) vi4*)(g),
      (__attribute__((address_space(3))) vi4*)(l), 0, 0);
#else
  // VGLOBAL async encoding: VDST = LDS byte address, VADDR = 64-bit global addr.
  unsigned lofs = (unsigned)(unsigned long long)(uintptr_t)l;   // addr[31:0] = LDS offset
  unsigned long long ga = (unsigned long long)(uintptr_t)g;
  asm volatile("global_load_async_to_lds_b128 %0, %1, off"
               :: "v"(lofs), "v"(ga) : "memory");
#endif
}

DEV void wait_async0() {
#if defined(__gfx1250__) && __has_builtin(__builtin_amdgcn_s_wait_asynccnt)
  __builtin_amdgcn_s_wait_asynccnt(0);
#else
  asm volatile("s_wait_asynccnt 0x0" ::: "memory");
#endif
}

// ---------------- WMMA helpers ----------------
DEV v8f wmma16(v16h a, v16h b, v8f c) {
  // D = A(16x32 f16) * B(32x16 f16) + C(16x16 f32)
  return __builtin_amdgcn_wmma_f32_16x16x32_f16(false, a, false, b, (short)0, c,
                                                false, false);
}

// A fragment (16x32) from row-major [m][k] f16 storage.
// ISA 7.12.2: lanes 0-15 / 16-31 both hold M=lane&15; lane-half selects K sub-blocks:
//   VGPR0-3 -> K = half*8 + 0..7 ; VGPR4-7 -> K = 16 + half*8 + 0..7
DEV v16h frag_a(const _Float16* p, int ld, int mbase, int kbase, int lane) {
  const _Float16* base =
      p + (size_t)(mbase + (lane & 15)) * ld + kbase + (lane >> 4) * 8;
  v8h lo = *(const v8h*)(base);
  v8h hi = *(const v8h*)(base + 16);
  v16h r;
#pragma unroll
  for (int i = 0; i < 8; ++i) { r[i] = lo[i]; r[i + 8] = hi[i]; }
  return r;
}

// B fragment (32x16) read from row-major B^T (== weight W[n][k]) f16 storage.
// lane holds column N=lane&15, K = half*8 + 0..7 and 16 + half*8 + 0..7 (matches frag_a).
DEV v16h frag_b(const _Float16* p, int ld, int nbase, int kbase, int lane) {
  const _Float16* base =
      p + (size_t)(nbase + (lane & 15)) * ld + kbase + (lane >> 4) * 8;
  v8h lo = *(const v8h*)(base);
  v8h hi = *(const v8h*)(base + 16);
  v16h r;
#pragma unroll
  for (int i = 0; i < 8; ++i) { r[i] = lo[i]; r[i + 8] = hi[i]; }
  return r;
}

// ---------------- K0: fp32 -> f16 weight conversion ----------------
__global__ void __launch_bounds__(256)
wcvt_kernel(const float* __restrict__ wq, const float* __restrict__ wk,
            const float* __restrict__ wv, const float* __restrict__ wm,
            const float* __restrict__ w1, const float* __restrict__ w2,
            unsigned char* __restrict__ ws) {
  int i = blockIdx.x * 256 + threadIdx.x;            // 0..40959
  if (i < 4096)       ((_Float16*)(ws + WQH))[i]         = (_Float16)wq[i];
  else if (i < 8192)  ((_Float16*)(ws + WKH))[i - 4096]  = (_Float16)wk[i - 4096];
  else if (i < 12288) ((_Float16*)(ws + WVH))[i - 8192]  = (_Float16)wv[i - 8192];
  else if (i < 16384) ((_Float16*)(ws + WMH))[i - 12288] = (_Float16)wm[i - 12288];
  else if (i < 32768) ((_Float16*)(ws + W1H))[i - 16384] = (_Float16)w1[i - 16384];
  else if (i < 40960) ((_Float16*)(ws + W2H))[i - 32768] = (_Float16)w2[i - 32768];
}

// ---------------- K0b: zero kv / ksum accumulators ----------------
__global__ void __launch_bounds__(256)
zero_kernel(unsigned char* __restrict__ ws) {
  int i = blockIdx.x * 256 + threadIdx.x;            // 133120 floats: kv then ksum
  if (i < 32 * 64 * 64 + 32 * 64) ((float*)(ws + KVO))[i] = 0.0f;
}

// ---------------- K1: q/k/v projections + feature map, f32 -> f16 ----------------
// 128 tokens per block, 8 waves: wave w owns M-tile w; 4 N-tiles, K=64 (2 WMMA steps).
__global__ void __launch_bounds__(256)
proj_kernel(const float* __restrict__ x, const float* __restrict__ src,
            unsigned char* __restrict__ ws) {
  __shared__ _Float16 sX[128 * 64];
  __shared__ _Float16 sS[128 * 64];
  const int tid = threadIdx.x, lane = tid & 31, wid = tid >> 5;
  const int t0 = blockIdx.x * 128;

  for (int i = tid; i < 128 * 64; i += 256) {
    sX[i] = (_Float16)x[(size_t)t0 * 64 + i];
    sS[i] = (_Float16)src[(size_t)t0 * 64 + i];
  }
  __builtin_prefetch((const _Float16*)(ws + WQH) + tid * 16, 0, 1);
  __syncthreads();

  const _Float16* wqh = (const _Float16*)(ws + WQH);
  const _Float16* wkh = (const _Float16*)(ws + WKH);
  const _Float16* wvh = (const _Float16*)(ws + WVH);
  _Float16* qf = (_Float16*)(ws + QFO);
  _Float16* kf = (_Float16*)(ws + KFO);
  _Float16* vf = (_Float16*)(ws + VFO);

  const int mt = wid;  // M-tile per wave
#pragma unroll
  for (int g = 0; g < 3; ++g) {
    const _Float16* A = (g == 0) ? sX : sS;
    const _Float16* W = (g == 0) ? wqh : (g == 1) ? wkh : wvh;
    _Float16* O = (g == 0) ? qf : (g == 1) ? kf : vf;
    // cache A fragments across N-tiles
    v16h a0 = frag_a(A, 64, mt * 16, 0, lane);
    v16h a1 = frag_a(A, 64, mt * 16, 32, lane);
#pragma unroll
    for (int nt = 0; nt < 4; ++nt) {
      v8f acc = {};
      acc = wmma16(a0, frag_b(W, 64, nt * 16, 0, lane), acc);
      acc = wmma16(a1, frag_b(W, 64, nt * 16, 32, lane), acc);
      const int col = nt * 16 + (lane & 15);
      const int rb  = mt * 16 + (lane >> 4) * 8;
#pragma unroll
      for (int i = 0; i < 8; ++i) {
        float v = acc[i];
        if (g < 2) v = (v > 0.0f) ? (v + 1.0f) : __expf(v);  // elu(v)+1
        else       v *= INV_S;                                // v / s_len
        O[(size_t)(t0 + rb + i) * 64 + col] = (_Float16)v;
      }
    }
  }
}

// ---------------- K2: kv[n,h] = K^T V  and  ksum[n,h] ----------------
// grid = 32 (n,h) * 8 S-chunks of 512. LDS holds transposed K^T / V^T tiles so both
// WMMA operands are contiguous 8-half (16B) reads. f32 atomic reduction to workspace.
__global__ void __launch_bounds__(256)
kv_kernel(unsigned char* __restrict__ ws) {
  __shared__ _Float16 sKT[64 * 128];  // [d][s]
  __shared__ _Float16 sVT[64 * 128];  // [e][s]
  const int tid = threadIdx.x, lane = tid & 31, wid = tid >> 5;
  const int nh = blockIdx.x >> 3, chunk = blockIdx.x & 7;
  const int n = nh >> 3, h = nh & 7;
  const _Float16* kf = (const _Float16*)(ws + KFO);
  const _Float16* vf = (const _Float16*)(ws + VFO);
  float* kv   = (float*)(ws + KVO) + (size_t)nh * 64 * 64;
  float* ksum = (float*)(ws + KSO) + (size_t)nh * 64;

  v8f acc0 = {}, acc1 = {};
  float ks = 0.0f;
  const int sBase = chunk * 512;

  for (int s0 = 0; s0 < 512; s0 += 128) {
    __syncthreads();
    for (int i = tid; i < 128 * 64; i += 256) {     // stage transposed
      const int sl = i >> 6, d = i & 63;
      const size_t t = ((size_t)(n * SSQ + sBase + s0 + sl) * HH + h);
      sKT[d * 128 + sl] = kf[t * 64 + d];
      sVT[d * 128 + sl] = vf[t * 64 + d];
    }
    __syncthreads();
    if (tid < 64) {                                  // ksum partial (row of sKT)
#pragma unroll 4
      for (int sl = 0; sl < 128; ++sl) ks += (float)sKT[tid * 128 + sl];
    }
    const int id0 = wid, id1 = wid + 8;              // 2 of 16 C tiles per wave
#pragma unroll
    for (int k = 0; k < 128; k += 32) {
      acc0 = wmma16(frag_a(sKT, 128, (id0 >> 2) * 16, k, lane),
                    frag_b(sVT, 128, (id0 & 3) * 16, k, lane), acc0);
      acc1 = wmma16(frag_a(sKT, 128, (id1 >> 2) * 16, k, lane),
                    frag_b(sVT, 128, (id1 & 3) * 16, k, lane), acc1);
    }
  }
  if (tid < 64) atomicAdd(&ksum[tid], ks);
  const int e0 = (lane & 15), rb = (lane >> 4) * 8;
#pragma unroll
  for (int i = 0; i < 8; ++i) {
    atomicAdd(&kv[((wid >> 2) * 16 + rb + i) * 64 + (wid & 3) * 16 + e0], acc0[i]);
    atomicAdd(&kv[(((wid + 8) >> 2) * 16 + rb + i) * 64 + ((wid + 8) & 3) * 16 + e0], acc1[i]);
  }
}

// ---------------- K3: fused  q@kv -> z -> wmerge -> LN1 -> [x|m]@w1 -> relu
//                               -> @w2 -> LN2 -> +x  ----------------
// One block = one (n,h) and 128 consecutive l. 64KB LDS, phase-aliased.
__global__ void __launch_bounds__(256)
fused_kernel(const float* __restrict__ x,
             const float* __restrict__ g1, const float* __restrict__ b1,
             const float* __restrict__ g2, const float* __restrict__ b2,
             float* __restrict__ out, unsigned char* __restrict__ ws) {
  __shared__ unsigned char smem[65536];
  _Float16* sA    = (_Float16*)smem;                         // [128][128]
  _Float16* kvT   = (_Float16*)(smem + 32768);               // [64][64]   (phase A)
  float*    ksumL = (float*)(smem + 32768 + 8192);           // [64]       (phase A)
  float*    zL    = (float*)(smem + 32768 + 8192 + 256);     // [128]      (phase A)
  float*    sF    = (float*)(smem + 32768);                  // [128][64]  (phase B alias)

  const int tid = threadIdx.x, lane = tid & 31, wid = tid >> 5;
  const int nh = blockIdx.x >> 5, lt = blockIdx.x & 31;
  const int n = nh >> 3, h = nh & 7, l0 = lt * 128;
  const _Float16* qf  = (const _Float16*)(ws + QFO);
  const _Float16* wmh = (const _Float16*)(ws + WMH);
  const _Float16* w1h = (const _Float16*)(ws + W1H);
  const _Float16* w2h = (const _Float16*)(ws + W2H);
  const float* kvG = (const float*)(ws + KVO) + (size_t)nh * 64 * 64;
  const float* ksG = (const float*)(ws + KSO) + (size_t)nh * 64;

  __builtin_prefetch(w1h + tid * 64, 0, 1);   // global_prefetch_b8: FFN weights -> L2
  __builtin_prefetch(w2h + tid * 32, 0, 1);

  // ---- phase 0: async DMA of the q' tile into sA cols 0..63 (ASYNCcnt path),
  //               overlapped with VALU-staged kv^T transpose + ksum ----
#pragma unroll
  for (int it = 0; it < 4; ++it) {
    const int idx = tid + it * 256;                 // 0..1023 lane-copies of 16B
    const int r = idx >> 3, seg = idx & 7;          // row, 16B segment within row
    const size_t t = ((size_t)(n * LL + l0 + r) * HH + h);
    async_cp_b128(qf + t * 64 + seg * 8, sA + r * 128 + seg * 8);
  }
  for (int i = tid; i < 4096; i += 256) {
    const int d = i >> 6, e = i & 63;
    kvT[e * 64 + d] = (_Float16)kvG[d * 64 + e];   // transpose so B-frag is contiguous
  }
  if (tid < 64) ksumL[tid] = ksG[tid];
  wait_async0();                                    // ASYNCcnt==0 before the barrier
  __syncthreads();

  // ---- phase 1: z[l] = S / (q'.ksum + eps) ----
  if (tid < 128) {
    float dot = 0.0f;
#pragma unroll 4
    for (int e = 0; e < 64; ++e) dot += (float)sA[tid * 128 + e] * ksumL[e];
    zL[tid] = (float)SSQ / (dot + EPS_ATTN);
  }
  __syncthreads();

  // ---- phase 2: GEMM1 msg = (q' @ kv) * z  -> f16 into sA cols 64..127 ----
  {
    v16h a0 = frag_a(sA, 128, wid * 16, 0, lane);
    v16h a1 = frag_a(sA, 128, wid * 16, 32, lane);
#pragma unroll
    for (int nt = 0; nt < 4; ++nt) {
      v8f acc = {};
      acc = wmma16(a0, frag_b(kvT, 64, nt * 16, 0, lane), acc);
      acc = wmma16(a1, frag_b(kvT, 64, nt * 16, 32, lane), acc);
      const int col = 64 + nt * 16 + (lane & 15);
      const int rb  = wid * 16 + (lane >> 4) * 8;
#pragma unroll
      for (int i = 0; i < 8; ++i)
        sA[(rb + i) * 128 + col] = (_Float16)(acc[i] * zL[rb + i]);
    }
  }
  __syncthreads();

  // ---- phase 3: GEMM2 msg2 = msg @ wmerge^T -> f32 sF ----
  {
    v16h a0 = frag_a(sA + 64, 128, wid * 16, 0, lane);
    v16h a1 = frag_a(sA + 64, 128, wid * 16, 32, lane);
#pragma unroll
    for (int nt = 0; nt < 4; ++nt) {
      v8f acc = {};
      acc = wmma16(a0, frag_b(wmh, 64, nt * 16, 0, lane), acc);
      acc = wmma16(a1, frag_b(wmh, 64, nt * 16, 32, lane), acc);
      const int col = nt * 16 + (lane & 15);
      const int rb  = wid * 16 + (lane >> 4) * 8;
#pragma unroll
      for (int i = 0; i < 8; ++i) sF[(rb + i) * 64 + col] = acc[i];
    }
  }
  __syncthreads();

  // ---- phase 4: LN1 -> sA cols 64..127 ; restage x (f16) -> sA cols 0..63 ----
  for (int i = tid; i < 8192; i += 256) {
    const int r = i >> 6, e = i & 63;
    const size_t t = ((size_t)(n * LL + l0 + r) * HH + h);
    sA[r * 128 + e] = (_Float16)x[t * 64 + e];
  }
  if (tid < 128) {
    float m = 0.0f;
#pragma unroll 4
    for (int e = 0; e < 64; ++e) m += sF[tid * 64 + e];
    m *= (1.0f / 64.0f);
    float v = 0.0f;
#pragma unroll 4
    for (int e = 0; e < 64; ++e) { float d = sF[tid * 64 + e] - m; v += d * d; }
    const float rs = rsqrtf(v * (1.0f / 64.0f) + EPS_LN);
#pragma unroll 4
    for (int e = 0; e < 64; ++e)
      sA[tid * 128 + 64 + e] = (_Float16)((sF[tid * 64 + e] - m) * rs * g1[e] + b1[e]);
  }
  __syncthreads();

  // ---- phase 5: GEMM3 hmid = relu(hcat @ w1^T), 128x128, K=128; in-place f16 ----
  {
    v16h a[4];
#pragma unroll
    for (int kk = 0; kk < 4; ++kk) a[kk] = frag_a(sA, 128, wid * 16, kk * 32, lane);
    v8f acc[8];
#pragma unroll
    for (int nt = 0; nt < 8; ++nt) {
      acc[nt] = (v8f){};
#pragma unroll
      for (int kk = 0; kk < 4; ++kk)
        acc[nt] = wmma16(a[kk], frag_b(w1h, 128, nt * 16, kk * 32, lane), acc[nt]);
    }
    // wave writes only its own rows; LDS ops from one wave are in-order -> safe
    const int rb = wid * 16 + (lane >> 4) * 8;
#pragma unroll
    for (int nt = 0; nt < 8; ++nt) {
      const int col = nt * 16 + (lane & 15);
#pragma unroll
      for (int i = 0; i < 8; ++i) {
        float v = acc[nt][i];
        sA[(rb + i) * 128 + col] = (_Float16)(v > 0.0f ? v : 0.0f);
      }
    }
  }

  // ---- phase 6: GEMM4 f = hmid @ w2^T -> f32 sF ----
  {
    v16h a[4];
#pragma unroll
    for (int kk = 0; kk < 4; ++kk) a[kk] = frag_a(sA, 128, wid * 16, kk * 32, lane);
#pragma unroll
    for (int nt = 0; nt < 4; ++nt) {
      v8f acc = {};
#pragma unroll
      for (int kk = 0; kk < 4; ++kk)
        acc = wmma16(a[kk], frag_b(w2h, 128, nt * 16, kk * 32, lane), acc);
      const int col = nt * 16 + (lane & 15);
      const int rb  = wid * 16 + (lane >> 4) * 8;
#pragma unroll
      for (int i = 0; i < 8; ++i) sF[(rb + i) * 64 + col] = acc[i];
    }
  }
  __syncthreads();

  // ---- phase 7: LN2 + residual -> out ----
  if (tid < 128) {
    float m = 0.0f;
#pragma unroll 4
    for (int e = 0; e < 64; ++e) m += sF[tid * 64 + e];
    m *= (1.0f / 64.0f);
    float v = 0.0f;
#pragma unroll 4
    for (int e = 0; e < 64; ++e) { float d = sF[tid * 64 + e] - m; v += d * d; }
    const float rs = rsqrtf(v * (1.0f / 64.0f) + EPS_LN);
    const size_t t = ((size_t)(n * LL + l0 + tid) * HH + h);
#pragma unroll 4
    for (int e = 0; e < 64; ++e)
      out[t * 64 + e] = x[t * 64 + e] + (sF[tid * 64 + e] - m) * rs * g2[e] + b2[e];
  }
}

// ---------------- host ----------------
extern "C" void kernel_launch(void* const* d_in, const int* in_sizes, int n_in,
                              void* d_out, int out_size, void* d_ws, size_t ws_size,
                              hipStream_t stream) {
  (void)in_sizes; (void)n_in; (void)out_size; (void)ws_size;
  const float* x   = (const float*)d_in[0];
  const float* src = (const float*)d_in[1];
  const float* wq  = (const float*)d_in[2];
  const float* wk  = (const float*)d_in[3];
  const float* wv  = (const float*)d_in[4];
  const float* wm  = (const float*)d_in[5];
  const float* w1  = (const float*)d_in[6];
  const float* w2  = (const float*)d_in[7];
  const float* g1  = (const float*)d_in[8];
  const float* b1  = (const float*)d_in[9];
  const float* g2  = (const float*)d_in[10];
  const float* b2  = (const float*)d_in[11];
  float* out = (float*)d_out;
  unsigned char* ws = (unsigned char*)d_ws;

  wcvt_kernel<<<160, 256, 0, stream>>>(wq, wk, wv, wm, w1, w2, ws);
  zero_kernel<<<520, 256, 0, stream>>>(ws);
  proj_kernel<<<TOK / 128, 256, 0, stream>>>(x, src, ws);
  kv_kernel<<<256, 256, 0, stream>>>(ws);
  fused_kernel<<<(NB * HH) * (LL / 128), 256, 0, stream>>>(x, g1, b1, g2, b2, out, ws);
}
